// DeepMarchingCube_80711025427112
// MI455X (gfx1250) — matrix operand order repaired and checked by reading, not compiled, not run on hardware
//
#include <hip/hip_runtime.h>
#include <cstdint>

typedef _Float16 h16;
typedef __attribute__((ext_vector_type(16))) _Float16 v16h;
typedef __attribute__((ext_vector_type(8)))  float    v8f;
typedef unsigned int u32;
typedef __attribute__((ext_vector_type(4))) u32 v4u;
typedef __attribute__((ext_vector_type(8))) int v8i;
typedef __attribute__((ext_vector_type(4))) int v4i;

#define TPB 256

// ---------------------------------------------------------------- utilities

__global__ void k_zero(float* __restrict__ p, long long n) {
  long long i = (long long)blockIdx.x * TPB + threadIdx.x;
  if (i < n) p[i] = 0.f;
}

__global__ void k_f2h(const float* __restrict__ s, h16* __restrict__ d, long long n) {
  long long i = (long long)blockIdx.x * TPB + threadIdx.x;
  if (i < n) d[i] = (h16)s[i];
}

// weights -> f16, padded [cout][tap][Cinp] layout (zeros for c >= Cin).
// flipT bit0 = flip taps, bit1 = src layout is (Cin,Cout,taps) (transposed conv)
__global__ void k_prep_w(const float* __restrict__ w, h16* __restrict__ o,
                         int Cout, int Cin, int Cinp, int flipT, long long n) {
  long long i = (long long)blockIdx.x * TPB + threadIdx.x;
  if (i >= n) return;
  int c   = (int)(i % Cinp);
  int tap = (int)((i / Cinp) % 27);
  int nn  = (int)(i / ((long long)Cinp * 27));
  h16 val = (h16)0;
  if (c < Cin) {
    int st = (flipT & 1) ? (26 - tap) : tap;
    long long sidx = (flipT & 2) ? (((long long)c * Cout + nn) * 27 + st)
                                 : (((long long)nn * Cin + c) * 27 + st);
    val = (h16)w[sidx];
  }
  o[i] = val;
}

// ------------------------------------------- point-feature MLP + voxelize (scatter max)

__global__ void k_fe(const float* __restrict__ x,
                     const float* __restrict__ w1, const float* __restrict__ b1,
                     const float* __restrict__ w2, const float* __restrict__ b2,
                     float* __restrict__ grid33, int nPts, int P) {
  int i = blockIdx.x * TPB + threadIdx.x;
  if (i >= nPts) return;
  int b = i / P;
  float x0 = x[3 * i], x1 = x[3 * i + 1], x2 = x[3 * i + 2];
  float h[16], f[16];
#pragma unroll
  for (int j = 0; j < 16; j++)
    h[j] = fmaxf(0.f, x0 * w1[j] + x1 * w1[16 + j] + x2 * w1[32 + j] + b1[j]);
  for (int j = 0; j < 16; j++) {
    float s = b2[j];
#pragma unroll
    for (int t = 0; t < 16; t++) s += h[t] * w2[t * 16 + j];
    f[j] = fmaxf(0.f, s);
  }
  int c0 = min(31, max(0, (int)floorf(x0 * 32.f)));
  int c1 = min(31, max(0, (int)floorf(x1 * 32.f)));
  int c2 = min(31, max(0, (int)floorf(x2 * 32.f)));
  int sp = (c0 * 33 + c1) * 33 + c2;
  long long base = (long long)b * 16 * 35937;
  for (int j = 0; j < 16; j++)  // relu => non-negative: int compare == float compare
    atomicMax((int*)grid33 + base + (long long)j * 35937 + sp, __float_as_int(f[j]));
}

// ---------------------------------------------------------------- WMMA implicit-GEMM conv3d
// 1 wave/block: 16 flat output voxels (M) x 16 output channels (N).
// K = 27 taps x Cinp channels (Cinp = max(Cin,32), weight tensor zero-padded),
// so each 32-wide K chunk lies inside a single tap: all spatial math hoisted per tap.
// B (weight) 16x32 tile is DMA'd to LDS by the Tensor Data Mover each chunk.

__global__ void __launch_bounds__(32)
k_conv_wmma(const h16* __restrict__ in, const h16* __restrict__ wt,
            const float* __restrict__ bias, float* __restrict__ out,
            int Cin, int Cinp, int Cout, int Din, int Dout, int pad,
            int physD, int off0) {
  __shared__ __align__(16) h16 ldsB[512];   // 16 rows x 32 halves = 1 KB

  const int lane = threadIdx.x;
  const int half = lane >> 4;
  const int mtile = blockIdx.x, ntile = blockIdx.y, b = blockIdx.z;
  const int Dout2 = Dout * Dout;
  const int M = Dout2 * Dout;
  const int m = mtile * 16 + (lane & 15);
  int od = 0, oh = 0, ow = 0;
  const bool mvalid = (m < M);
  if (mvalid) { ow = m % Dout; oh = (m / Dout) % Dout; od = m / Dout2; }
  const int n = ntile * 16 + (lane & 15);
  const int physD2 = physD * physD;
  const int physD3 = physD2 * physD;
  const int inBase = b * Cin * physD3;      // all activation indices fit in 32 bits
  const int Ktotp = 27 * Cinp;

  // lane-constant A fragment element offsets: klocal = i + half*16
  int  aOff[16];
  bool cok[16];
#pragma unroll
  for (int i = 0; i < 16; i++) {
    int klo = i + (half << 4);
    aOff[i] = klo * physD3;
    cok[i]  = (klo < Cin);                  // only false on Cin==16 layers
  }

  v8f acc = {};
  for (int tap = 0; tap < 27; ++tap) {
    int kd = tap / 9, r = tap - kd * 9, kh = r / 3, kw = r - kh * 3;
    int id = od + kd - pad, ih = oh + kh - pad, iw = ow + kw - pad;
    bool tapOK = mvalid && (unsigned)id < (unsigned)Din &&
                 (unsigned)ih < (unsigned)Din && (unsigned)iw < (unsigned)Din;
    int aTap = inBase + (id + off0) * physD2 + (ih + off0) * physD + (iw + off0);
    __builtin_prefetch(in + aTap, 0, 1);    // global_prefetch_b8

    for (int cb = 0; cb < Cinp; cb += 32) {
      int x0 = tap * Cinp + cb;             // within-row K offset (elements)

#if __has_builtin(__builtin_amdgcn_tensor_load_to_lds) && __has_builtin(__builtin_amdgcn_s_wait_tensorcnt)
      // ---- TDM: 2D tile, 16 rows x 32 halves, row stride 27*Cinp, OOB rows -> 0
      unsigned long long ga = (unsigned long long)(uintptr_t)wt +
                              ((unsigned long long)(ntile * 16) * Ktotp + x0) * 2ull;
      u32 lds_addr = (u32)(uintptr_t)ldsB;
      u32 dim0 = (u32)(Ktotp - x0);                         // remaining elems in row
      u32 dim1 = (u32)max(0, Cout - ntile * 16);            // valid rows
      v4u g0;
      g0[0] = 1u;                                           // count=1, user mode
      g0[1] = lds_addr;                                     // lds_addr [63:32]
      g0[2] = (u32)ga;                                      // global_addr lo
      g0[3] = ((u32)(ga >> 32) & 0x01FFFFFFu) | (2u << 30); // addr[56:32] | type=2
      v8i g1;
      g1[0] = (int)(1u << 16);                              // data_size=2B
      g1[1] = (int)((dim0 & 0xFFFFu) << 16);                // tensor_dim0[15:0]
      g1[2] = (int)(((dim0 >> 16) & 0xFFFFu) | ((dim1 & 0xFFFFu) << 16));
      g1[3] = (int)(((dim1 >> 16) & 0xFFFFu) | (32u << 16));// tile_dim0=32
      g1[4] = 16;                                           // tile_dim1=16, tile_dim2=0
      g1[5] = (int)(u32)(27 * Cinp);                        // tensor_dim0_stride lo32
      g1[6] = 0;                                            // stride hi | dim1_stride lo
      g1[7] = 0;
      v4i gz4 = {0, 0, 0, 0};
      v8i gz8 = {0, 0, 0, 0, 0, 0, 0, 0};
      __builtin_amdgcn_tensor_load_to_lds(g0, g1, gz4, gz4, gz8, 0);
#endif

      // ---- A fragment: per-lane gathers, one shared base per chunk
      int aBase = aTap + cb * physD3;
      v16h av;
#pragma unroll
      for (int i = 0; i < 16; i++)
        av[i] = (tapOK && cok[i]) ? in[aBase + aOff[i]] : (h16)0;

      // ---- B fragment
      v16h bv;
#if __has_builtin(__builtin_amdgcn_tensor_load_to_lds) && __has_builtin(__builtin_amdgcn_s_wait_tensorcnt)
      __builtin_amdgcn_s_wait_tensorcnt(0);
      asm volatile("" ::: "memory");
      {
        union { v16h v; uint4 q[2]; } bu;
        const uint4* lp = (const uint4*)ldsB + (lane & 15) * 4 + half * 2;
        bu.q[0] = lp[0];
        bu.q[1] = lp[1];
        bv = bu.v;
      }
#else
      {
        union { v16h v; uint4 q[2]; } bu;
        if (n < Cout) {
          const uint4* bp = (const uint4*)(wt + (long long)n * Ktotp + x0 + (half << 4));
          bu.q[0] = bp[0];
          bu.q[1] = bp[1];
        } else {
          bu.q[0] = make_uint4(0, 0, 0, 0);
          bu.q[1] = make_uint4(0, 0, 0, 0);
        }
        bv = bu.v;
      }
#endif
      acc = __builtin_amdgcn_wmma_f32_16x16x32_f16(false, av, false, bv,
                                                   (short)0, acc, false, false);
    }
  }

  if (n < Cout) {
    float bb = bias[n];
    long long outBase = ((long long)b * Cout + n) * M;
#pragma unroll
    for (int r = 0; r < 8; r++) {
      int mg = mtile * 16 + r + half * 8;     // C/D layout: M = r + half*8
      if (mg < M) out[outBase + mg] = acc[r] + bb;
    }
  }
}

// ---------------------------------------------------------------- BatchNorm

__global__ void k_bn_stats(const float* __restrict__ x, float* __restrict__ stats,
                           int C, long long S, int B) {
  int c = blockIdx.x, t = threadIdx.x;
  __shared__ float ss[TPB];
  __shared__ float sq[TPB];
  float s = 0.f, q = 0.f;
  long long per = (long long)B * S;
  for (long long j = t; j < per; j += TPB) {
    long long bb = j / S;
    long long sp = j - bb * S;
    float v = x[((long long)bb * C + c) * S + sp];
    s += v; q += v * v;
  }
  ss[t] = s; sq[t] = q;
  __syncthreads();
  for (int w = TPB / 2; w > 0; w >>= 1) {
    if (t < w) { ss[t] += ss[t + w]; sq[t] += sq[t + w]; }
    __syncthreads();
  }
  if (t == 0) { stats[2 * c] = ss[0]; stats[2 * c + 1] = sq[0]; }
}

__global__ void k_bn_relu(float* __restrict__ x, const float* __restrict__ stats,
                          const float* __restrict__ g, const float* __restrict__ bb,
                          int C, long long S, long long total, float invCnt) {
  long long i = (long long)blockIdx.x * TPB + threadIdx.x;
  if (i >= total) return;
  int c = (int)((i / S) % C);
  float m = stats[2 * c] * invCnt;
  float v = stats[2 * c + 1] * invCnt - m * m;
  float y = (x[i] - m) * rsqrtf(v + 1e-5f) * g[c] + bb[c];
  x[i] = fmaxf(y, 0.f);
}

// ---------------------------------------------------------------- pool / unpool / concat

__global__ void k_maxpool(const float* __restrict__ in, float* __restrict__ v,
                          int* __restrict__ idx, int C, int D, int Dp, long long total) {
  long long i = (long long)blockIdx.x * TPB + threadIdx.x;
  if (i >= total) return;
  int ow = (int)(i % Dp); long long t = i / Dp;
  int oh = (int)(t % Dp); t /= Dp;
  int od = (int)(t % Dp); t /= Dp;
  int c  = (int)(t % C);
  int b  = (int)(t / C);
  const float* base = in + ((long long)b * C + c) * D * D * D;
  float best = -3.4e38f; int bk = 0;
#pragma unroll
  for (int kd = 0; kd < 2; kd++)
#pragma unroll
    for (int kh = 0; kh < 2; kh++)
#pragma unroll
      for (int kw = 0; kw < 2; kw++) {
        float val = base[((long long)(2 * od + kd) * D + (2 * oh + kh)) * D + (2 * ow + kw)];
        int kk = kd * 4 + kh * 2 + kw;
        if (val > best) { best = val; bk = kk; }
      }
  v[i] = best;
  int kd = bk >> 2, kh = (bk >> 1) & 1, kw = bk & 1;
  idx[i] = ((2 * od + kd) * D + (2 * oh + kh)) * D + (2 * ow + kw);
}

__global__ void k_unpool(const float* __restrict__ v, const int* __restrict__ idx,
                         float* __restrict__ dst, long long Sp, long long S, long long total) {
  long long i = (long long)blockIdx.x * TPB + threadIdx.x;
  if (i >= total) return;
  long long bc = i / Sp;
  dst[bc * S + idx[i]] = v[i];
}

__global__ void k_concat(float* __restrict__ dst, const float* __restrict__ a,
                         const float* __restrict__ bsrc, int Ca, int Cb,
                         long long S, long long total) {
  long long i = (long long)blockIdx.x * TPB + threadIdx.x;
  if (i >= total) return;
  long long s = i % S;
  long long t = i / S;
  int c = (int)(t % (Ca + Cb));
  int b = (int)(t / (Ca + Cb));
  dst[i] = (c < Ca) ? a[((long long)b * Ca + c) * S + s]
                    : bsrc[((long long)b * Cb + (c - Ca)) * S + s];
}

__global__ void k_sigmoid(const float* __restrict__ x, float* __restrict__ y,
                          float addc, long long n) {
  long long i = (long long)blockIdx.x * TPB + threadIdx.x;
  if (i >= n) return;
  y[i] = 1.f / (1.f + expf(-x[i])) + addc;
}

// ---------------------------------------------------------------- marching-cube table

__global__ void k_top(const float* __restrict__ occ, float* __restrict__ top, long long total) {
  long long i = (long long)blockIdx.x * TPB + threadIdx.x;
  if (i >= total) return;
  int cell = (int)(i & 32767);
  int b    = (int)(i >> 15);
  int cz = cell & 31, cy = (cell >> 5) & 31, cx = cell >> 10;
  float p[8];
#pragma unroll
  for (int di = 0; di < 2; di++)
#pragma unroll
    for (int dj = 0; dj < 2; dj++)
#pragma unroll
      for (int dk = 0; dk < 2; dk++)
        p[di * 4 + dj * 2 + dk] =
            occ[(((long long)b * 33 + cx + di) * 33 + (cy + dj)) * 33 + (cz + dk)];
  float* o = top + i * 256;
  for (int code = 0; code < 256; code++) {
    float prod = 1.f;
#pragma unroll
    for (int k = 0; k < 8; k++) {
      float pk = p[k];
      prod *= ((code >> k) & 1) ? pk : (1.f - pk);
    }
    o[code] = prod;
  }
}

// ---------------------------------------------------------------- host orchestration

extern "C" void kernel_launch(void* const* d_in, const int* in_sizes, int n_in,
                              void* d_out, int out_size, void* d_ws, size_t ws_size,
                              hipStream_t stream) {
  auto P = [&](int i) -> const float* { return (const float*)d_in[i]; };
  size_t off_ = 0;
  auto alloc = [&](size_t bytes) -> void* {
    void* p = (char*)d_ws + off_;
    off_ += (bytes + 255) & ~(size_t)255;
    return p;
  };
  auto nb = [](long long n) { return dim3((unsigned)((n + TPB - 1) / TPB)); };

  const int B = 8;
  const long long S33 = 35937, S37 = 50653, S18 = 5832, S9 = 729, S4 = 64;

  auto zero = [&](float* p, long long n) { k_zero<<<nb(n), TPB, 0, stream>>>(p, n); };

  auto conv = [&](const float* inF, int Cin, int Cout, int Din, int pad,
                  int physD, int off0, int Dout, int wIdx, int flipT,
                  const float* bias, float* outF) {
    long long inE = (long long)B * Cin * physD * physD * physD;
    h16* inH = (h16*)alloc(inE * 2);
    k_f2h<<<nb(inE), TPB, 0, stream>>>(inF, inH, inE);
    int Cinp = (Cin < 32) ? 32 : Cin;
    long long wE = (long long)Cout * 27 * Cinp;
    h16* wH = (h16*)alloc(wE * 2);
    k_prep_w<<<nb(wE), TPB, 0, stream>>>(P(wIdx), wH, Cout, Cin, Cinp, flipT, wE);
    long long M = (long long)Dout * Dout * Dout;
    dim3 g((unsigned)((M + 15) / 16), (unsigned)((Cout + 15) / 16), B);
    k_conv_wmma<<<g, 32, 0, stream>>>(inH, wH, bias, outF, Cin, Cinp, Cout,
                                      Din, Dout, pad, physD, off0);
  };
  auto bnrelu = [&](float* x, int C, long long S, int gI, int bI) {
    float* st = (float*)alloc(2 * C * sizeof(float));
    k_bn_stats<<<C, TPB, 0, stream>>>(x, st, C, S, B);
    long long tot = (long long)B * C * S;
    k_bn_relu<<<nb(tot), TPB, 0, stream>>>(x, st, P(gI), P(bI), C, S, tot,
                                           1.f / (float)(B * S));
  };

  // ---- point features -> 33^3 padded grid (16 ch)
  float* g33 = (float*)alloc((size_t)B * 16 * S33 * 4);
  zero(g33, (long long)B * 16 * S33);
  k_fe<<<nb(24000), TPB, 0, stream>>>(P(0), P(1), P(2), P(3), P(4), g33, 24000, 3000);

  // ---- encoder
  float* e1 = (float*)alloc((size_t)B * 16 * S37 * 4);
  conv(g33, 16, 16, 33, 3, 33, 0, 37, 5, 0, P(6), e1);  bnrelu(e1, 16, S37, 7, 8);
  float* e2 = (float*)alloc((size_t)B * 16 * S37 * 4);
  conv(e1, 16, 16, 37, 1, 37, 0, 37, 9, 0, P(10), e2);  bnrelu(e2, 16, S37, 11, 12);

  float* p1v = (float*)alloc((size_t)B * 16 * S18 * 4);
  int*   p1i = (int*)alloc((size_t)B * 16 * S18 * 4);
  k_maxpool<<<nb((long long)B * 16 * S18), TPB, 0, stream>>>(e2, p1v, p1i, 16, 37, 18,
                                                             (long long)B * 16 * S18);
  float* c21 = (float*)alloc((size_t)B * 32 * S18 * 4);
  conv(p1v, 16, 32, 18, 1, 18, 0, 18, 13, 0, P(14), c21); bnrelu(c21, 32, S18, 15, 16);
  float* c22 = (float*)alloc((size_t)B * 32 * S18 * 4);
  conv(c21, 32, 32, 18, 1, 18, 0, 18, 17, 0, P(18), c22); bnrelu(c22, 32, S18, 19, 20);

  float* p2v = (float*)alloc((size_t)B * 32 * S9 * 4);
  int*   p2i = (int*)alloc((size_t)B * 32 * S9 * 4);
  k_maxpool<<<nb((long long)B * 32 * S9), TPB, 0, stream>>>(c22, p2v, p2i, 32, 18, 9,
                                                            (long long)B * 32 * S9);
  float* c31 = (float*)alloc((size_t)B * 64 * S9 * 4);
  conv(p2v, 32, 64, 9, 1, 9, 0, 9, 21, 0, P(22), c31); bnrelu(c31, 64, S9, 23, 24);
  float* c32 = (float*)alloc((size_t)B * 64 * S9 * 4);
  conv(c31, 64, 64, 9, 1, 9, 0, 9, 25, 0, P(26), c32); bnrelu(c32, 64, S9, 27, 28);

  float* p3v = (float*)alloc((size_t)B * 64 * S4 * 4);
  int*   p3i = (int*)alloc((size_t)B * 64 * S4 * 4);
  k_maxpool<<<nb((long long)B * 64 * S4), TPB, 0, stream>>>(c32, p3v, p3i, 64, 9, 4,
                                                            (long long)B * 64 * S4);
  float* c4 = (float*)alloc((size_t)B * 128 * S4 * 4);
  conv(p3v, 64, 128, 4, 1, 4, 0, 4, 29, 0, P(30), c4); bnrelu(c4, 128, S4, 31, 32);
  float* d4 = (float*)alloc((size_t)B * 64 * S4 * 4);
  conv(c4, 128, 64, 4, 1, 4, 0, 4, 33, 0, P(34), d4);  bnrelu(d4, 64, S4, 35, 36);

  // ---- decoder
  float* u3 = (float*)alloc((size_t)B * 64 * S9 * 4);
  zero(u3, (long long)B * 64 * S9);
  k_unpool<<<nb((long long)B * 64 * S4), TPB, 0, stream>>>(d4, p3i, u3, S4, S9,
                                                           (long long)B * 64 * S4);
  float* cc3 = (float*)alloc((size_t)B * 128 * S9 * 4);
  k_concat<<<nb((long long)B * 128 * S9), TPB, 0, stream>>>(cc3, c32, u3, 64, 64, S9,
                                                            (long long)B * 128 * S9);
  float* d31 = (float*)alloc((size_t)B * 128 * S9 * 4);
  conv(cc3, 128, 128, 9, 1, 9, 0, 9, 37, 3, P(38), d31); bnrelu(d31, 128, S9, 39, 40);
  float* d32 = (float*)alloc((size_t)B * 32 * S9 * 4);
  conv(d31, 128, 32, 9, 1, 9, 0, 9, 41, 3, P(42), d32);  bnrelu(d32, 32, S9, 43, 44);

  float* u2 = (float*)alloc((size_t)B * 32 * S18 * 4);
  zero(u2, (long long)B * 32 * S18);
  k_unpool<<<nb((long long)B * 32 * S9), TPB, 0, stream>>>(d32, p2i, u2, S9, S18,
                                                           (long long)B * 32 * S9);
  float* cc2 = (float*)alloc((size_t)B * 64 * S18 * 4);
  k_concat<<<nb((long long)B * 64 * S18), TPB, 0, stream>>>(cc2, c22, u2, 32, 32, S18,
                                                            (long long)B * 64 * S18);
  // occ / off branches (18^3)
  float* q1 = (float*)alloc((size_t)B * 64 * S18 * 4);
  conv(cc2, 64, 64, 18, 1, 18, 0, 18, 45, 3, P(46), q1); bnrelu(q1, 64, S18, 47, 48);
  float* q2 = (float*)alloc((size_t)B * 16 * S18 * 4);
  conv(q1, 64, 16, 18, 1, 18, 0, 18, 49, 3, P(50), q2);  bnrelu(q2, 16, S18, 51, 52);
  float* r1 = (float*)alloc((size_t)B * 64 * S18 * 4);
  conv(cc2, 64, 64, 18, 1, 18, 0, 18, 53, 3, P(54), r1); bnrelu(r1, 64, S18, 55, 56);
  float* r2 = (float*)alloc((size_t)B * 16 * S18 * 4);
  conv(r1, 64, 16, 18, 1, 18, 0, 18, 57, 3, P(58), r2);  bnrelu(r2, 16, S18, 59, 60);

  float* uo = (float*)alloc((size_t)B * 16 * S37 * 4);
  zero(uo, (long long)B * 16 * S37);
  k_unpool<<<nb((long long)B * 16 * S18), TPB, 0, stream>>>(q2, p1i, uo, S18, S37,
                                                            (long long)B * 16 * S18);
  float* uf = (float*)alloc((size_t)B * 16 * S37 * 4);
  zero(uf, (long long)B * 16 * S37);
  k_unpool<<<nb((long long)B * 16 * S18), TPB, 0, stream>>>(r2, p1i, uf, S18, S37,
                                                            (long long)B * 16 * S18);
  float* co = (float*)alloc((size_t)B * 32 * S37 * 4);
  k_concat<<<nb((long long)B * 32 * S37), TPB, 0, stream>>>(co, e2, uo, 16, 16, S37,
                                                            (long long)B * 32 * S37);
  float* cf = (float*)alloc((size_t)B * 32 * S37 * 4);
  k_concat<<<nb((long long)B * 32 * S37), TPB, 0, stream>>>(cf, e2, uf, 16, 16, S37,
                                                            (long long)B * 32 * S37);

  float* o1 = (float*)alloc((size_t)B * 32 * S37 * 4);
  conv(co, 32, 32, 37, 1, 37, 0, 37, 61, 3, P(62), o1); bnrelu(o1, 32, S37, 63, 64);
  float* f1 = (float*)alloc((size_t)B * 32 * S37 * 4);
  conv(cf, 32, 32, 37, 1, 37, 0, 37, 67, 3, P(68), f1); bnrelu(f1, 32, S37, 69, 70);

  // final transposed convs (pad=3 -> crop 1 each side: logical 35^3 view of 37^3, conv pad0 -> 33^3)
  float* o2 = (float*)alloc((size_t)B * 1 * S33 * 4);
  conv(o1, 32, 1, 35, 0, 37, 1, 33, 65, 3, P(66), o2);
  float* f2 = (float*)alloc((size_t)B * 3 * S33 * 4);
  conv(f1, 32, 3, 35, 0, 37, 1, 33, 71, 3, P(72), f2);

  // ---- outputs: (off, top, occ) concatenated
  float* out = (float*)d_out;
  long long offN = (long long)B * 3 * S33;
  long long topN = (long long)B * 32768 * 256;
  float* outOff = out;
  float* outTop = out + offN;
  float* outOcc = out + offN + topN;
  k_sigmoid<<<nb(offN), TPB, 0, stream>>>(f2, outOff, -0.5f, offN);
  k_sigmoid<<<nb((long long)B * S33), TPB, 0, stream>>>(o2, outOcc, 0.f, (long long)B * S33);
  k_top<<<nb((long long)B * 32768), TPB, 0, stream>>>(outOcc, outTop, (long long)B * 32768);
}
